// T10_transformer_11751030522452
// MI455X (gfx1250) — matrix operand order repaired and checked by reading, not compiled, not run on hardware
//
#include <hip/hip_runtime.h>
#include <cstdint>

// ---------------- problem constants ----------------
#define BS     64
#define SEQ    2048
#define LQ     128
#define NHEADS 4
#define DT     128      // D_TIME
#define DK     32       // per-head dim
#define NH     128
#define REGH   200
#define KCHUNK 256      // keys staged per TDM transfer in attention
#define KI_SCALE 2.3873241463784303f     // 1 / radians(24.0)
#define SCORE_SCALE 0.17677669529663687f // 1/sqrt(32)

typedef _Float16 v8h  __attribute__((ext_vector_type(8)));
typedef _Float16 v16h __attribute__((ext_vector_type(16)));
typedef float    v8f  __attribute__((ext_vector_type(8)));
typedef unsigned int u32x4 __attribute__((ext_vector_type(4)));
typedef int      i32x8 __attribute__((ext_vector_type(8)));
typedef int      i32x4 __attribute__((ext_vector_type(4)));

#if defined(__HIP_DEVICE_COMPILE__) && __has_builtin(__builtin_amdgcn_tensor_load_to_lds) && __has_builtin(__builtin_amdgcn_s_wait_tensorcnt)
#define USE_TDM 1
#else
#define USE_TDM 0
#endif

__device__ __forceinline__ v16h make16(v8h lo, v8h hi) {
    v16h r;
#pragma unroll
    for (int i = 0; i < 8; ++i) { r[i] = lo[i]; r[i + 8] = hi[i]; }
    return r;
}

__device__ __forceinline__ float sigmoidf_(float x) { return 1.0f / (1.0f + __expf(-x)); }
__device__ __forceinline__ float gelu_exact(float x) { return 0.5f * x * (1.0f + erff(x * 0.70710678118654752f)); }

#if USE_TDM
// Issue a TDM 2-D tile load: KCHUNK rows x 32 halfs (row stride DT halfs) -> LDS.
// clang-23 6-arg form: (g0 u32x4, g1 i32x8, g2 i32x4, g3 i32x4, i32x8, cpol)
__device__ __forceinline__ void tdm_load_k(const _Float16* gptr, unsigned int lds_off) {
    const unsigned long long ga = (unsigned long long)(uintptr_t)gptr;
    u32x4 g0;
    g0[0] = 1u;                                   // count=1 (valid user descriptor)
    g0[1] = lds_off;                              // LDS byte offset of tile start
    g0[2] = (unsigned int)ga;                     // global_addr[31:0]
    g0[3] = (unsigned int)((ga >> 32) & 0x01FFFFFFu) | (2u << 30); // addr[56:32] | type=2
    // group1: wg_mask=0 | data_size=2B ; tensor_dim0=32 ; tensor_dim1=SEQ ;
    //         tile_dim0=32 ; tile_dim1=KCHUNK ; tile_dim2=0 ; stride0=DT ; stride1=0
    const i32x8 g1 = { 0x00010000,            // [17:16] data_size=1 (2 bytes)
                       (32 << 16),            // tensor_dim0[15:0] << 16
                       (SEQ << 16),           // tensor_dim1 low16 << 16 (high16 -> next dw = 0)
                       (32 << 16),            // tile_dim0 << 16 (tensor_dim1 hi16 = 0)
                       KCHUNK,                // tile_dim1 (tile_dim2 = 0)
                       DT,                    // tensor_dim0_stride low32
                       0, 0 };                // stride0 hi16 / tensor_dim1_stride
    const i32x4 z4 = { 0, 0, 0, 0 };
    const i32x8 z8 = { 0, 0, 0, 0, 0, 0, 0, 0 };
    __builtin_amdgcn_tensor_load_to_lds(g0, g1, z4, z4, z8, 0);
}
#endif

// =====================================================================
// K0: q time-embed + q projection -> q16[h][q][dk] f16
// grid: 128 blocks (one per query), 128 threads
// =====================================================================
__global__ void qprep_kernel(const float* __restrict__ te_lin_w, const float* __restrict__ te_lin_b,
                             const float* __restrict__ te_per_w, const float* __restrict__ te_per_b,
                             const float* __restrict__ q_w, const float* __restrict__ q_b,
                             _Float16* __restrict__ q16) {
    __shared__ float femb[DT];
    const int qi = blockIdx.x;
    const int e  = threadIdx.x;
    const float t = (float)qi * (1.0f / 127.0f);
    femb[e] = (e == 0) ? (t * te_lin_w[0] + te_lin_b[0])
                       : __sinf(t * te_per_w[e - 1] + te_per_b[e - 1]);
    __syncthreads();
    const int d = threadIdx.x;
    float acc = q_b[d];
    for (int k = 0; k < DT; ++k) acc = fmaf(femb[k], q_w[d * DT + k], acc);
    q16[((size_t)((d >> 5) * LQ + qi)) * DK + (d & 31)] = (_Float16)acc;
}

// =====================================================================
// K1: fuse att_out projection into GRU input gates:
//     W2[g][c] = sum_j wih[g][j]*att_out_w[j][c],  b2[g]=sum_j wih[g][j]*aob[j]+bih[g]
// =====================================================================
__global__ void fusew2_kernel(const float* __restrict__ wih, const float* __restrict__ aow,
                              const float* __restrict__ aob, const float* __restrict__ bih,
                              float* __restrict__ W2, float* __restrict__ b2) {
    const int g = blockIdx.x * blockDim.x + threadIdx.x;
    if (g >= 3 * NH) return;
    float bacc = bih[g];
    float wacc[8];
#pragma unroll
    for (int c = 0; c < 8; ++c) wacc[c] = 0.0f;
    for (int j = 0; j < NH; ++j) {
        const float w = wih[(size_t)g * NH + j];
        bacc = fmaf(w, aob[j], bacc);
#pragma unroll
        for (int c = 0; c < 8; ++c) wacc[c] = fmaf(w, aow[j * 8 + c], wacc[c]);
    }
#pragma unroll
    for (int c = 0; c < 8; ++c) W2[g * 8 + c] = wacc[c];
    b2[g] = bacc;
}

// =====================================================================
// K2: key time-embed (sin) + WMMA projection  k16[b][s][d] (f16)
// grid: BS*SEQ/64 = 2048 blocks, 256 threads (8 waves)
// =====================================================================
__global__ void __launch_bounds__(256) kproj_kernel(const float* __restrict__ fa_vals,
                             const float* __restrict__ te_lin_w, const float* __restrict__ te_lin_b,
                             const float* __restrict__ te_per_w, const float* __restrict__ te_per_b,
                             const float* __restrict__ k_w, const float* __restrict__ k_b,
                             _Float16* __restrict__ k16) {
    __shared__ _Float16 semb[64][DT];
    __shared__ _Float16 skw[DT][DT];
    const int b  = blockIdx.x >> 5;
    const int s0 = (blockIdx.x & 31) * 64;
    const int tid = threadIdx.x;

    for (int i = tid; i < DT * DT; i += 256) skw[i >> 7][i & 127] = (_Float16)k_w[i];

    const float lw = te_lin_w[0], lb = te_lin_b[0];
    for (int i = tid; i < 64 * DT; i += 256) {
        const int row = i >> 7, e = i & 127;
        const float ki = fa_vals[(size_t)b * SEQ + s0 + row] * KI_SCALE;
        const float v = (e == 0) ? (ki * lw + lb)
                                 : __sinf(ki * te_per_w[e - 1] + te_per_b[e - 1]);
        semb[row][e] = (_Float16)v;
    }
    __syncthreads();

    const int wv = tid >> 5, lane = tid & 31;
    const int lm = lane & 15, hi = lane >> 4;
    const int mt = wv & 3;
    const int ntbase = (wv >> 2) * 4;
    for (int nt = ntbase; nt < ntbase + 4; ++nt) {
        v8f acc = {};
#pragma unroll
        for (int ks = 0; ks < 4; ++ks) {
            const int e0 = ks * 32;
            v8h a0 = *(const v8h*)&semb[mt * 16 + lm][e0 + 8 * hi];
            v8h a1 = *(const v8h*)&semb[mt * 16 + lm][e0 + 16 + 8 * hi];
            v8h b0 = *(const v8h*)&skw[nt * 16 + lm][e0 + 8 * hi];
            v8h b1 = *(const v8h*)&skw[nt * 16 + lm][e0 + 16 + 8 * hi];
            acc = __builtin_amdgcn_wmma_f32_16x16x32_f16(false, make16(a0, a1),
                                                         false, make16(b0, b1),
                                                         (short)0, acc, false, false);
        }
        const float kb = k_b[nt * 16 + lm];
#pragma unroll
        for (int r = 0; r < 8; ++r) {
            const int row = mt * 16 + r + 8 * hi;
            k16[((size_t)b * SEQ + s0 + row) * DT + nt * 16 + lm] = (_Float16)(acc[r] + kb);
        }
    }
}

// =====================================================================
// K3: flash attention — TDM-staged keys in LDS + WMMA scores + online softmax
// grid: 256 blocks = (b,h), 256 threads = 8 waves, wave = one 16-query tile
// =====================================================================
__global__ void __launch_bounds__(256) attn_kernel(const _Float16* __restrict__ q16,
                            const _Float16* __restrict__ k16,
                            const float* __restrict__ X_fa,
                            const unsigned char* __restrict__ fa_mask,
                            float* __restrict__ concat) {
#if USE_TDM
    __shared__ __align__(16) _Float16 kbuf[2][KCHUNK * DK];   // 2 x 16KB double buffer
#endif
    const int b  = blockIdx.x >> 2;
    const int hh = blockIdx.x & 3;
    const int wv = threadIdx.x >> 5, lane = threadIdx.x & 31;
    const int lm = lane & 15, hi = lane >> 4;
    const int qt = wv;

    const _Float16* qr = q16 + ((size_t)(hh * LQ + qt * 16 + lm)) * DK;
    const v16h amat = make16(*(const v8h*)(qr + 16 * hi), *(const v8h*)(qr + 16 * hi + 8));

    float m[8], l[8], a0[8], a1[8];
#pragma unroll
    for (int r = 0; r < 8; ++r) { m[r] = -1e30f; l[r] = 0.0f; a0[r] = 0.0f; a1[r] = 0.0f; }

    const size_t kbase = (size_t)b * SEQ;

#if USE_TDM
    const int NCH = SEQ / KCHUNK;          // 8 chunks
    if (wv == 0)
        tdm_load_k(k16 + kbase * DT + hh * DK, (unsigned int)(uintptr_t)&kbuf[0][0]);
    for (int c = 0; c < NCH; ++c) {
        if (wv == 0) __builtin_amdgcn_s_wait_tensorcnt(0);  // chunk c landed in LDS
        __syncthreads();
        if (wv == 0 && c + 1 < NCH)
            tdm_load_k(k16 + (kbase + (size_t)(c + 1) * KCHUNK) * DT + hh * DK,
                       (unsigned int)(uintptr_t)&kbuf[(c + 1) & 1][0]);
        if (c + 1 < NCH)                                     // warm L2/WGP$ for values
            __builtin_prefetch(X_fa + kbase + (c + 1) * KCHUNK + lane * 8, 0, 3);
        const _Float16* kb = kbuf[c & 1];
        for (int t = 0; t < KCHUNK / 16; ++t) {
            const int s0 = c * KCHUNK + t * 16;
            const v16h bmat = make16(*(const v8h*)&kb[(t * 16 + lm) * DK + 16 * hi],
                                     *(const v8h*)&kb[(t * 16 + lm) * DK + 16 * hi + 8]);
#else
    {
        for (int s0 = 0; s0 < SEQ; s0 += 16) {
            const _Float16* kr = k16 + (kbase + s0 + lm) * DT + hh * DK + 16 * hi;
            const v16h bmat = make16(*(const v8h*)kr, *(const v8h*)(kr + 8));
#endif
            v8f sc = {};
            sc = __builtin_amdgcn_wmma_f32_16x16x32_f16(false, amat, false, bmat,
                                                        (short)0, sc, false, false);
            const int col = s0 + lm;
            const float xv = X_fa[kbase + col];
            const float mf = fa_mask[kbase + col] ? 1.0f : 0.0f;
#pragma unroll
            for (int r = 0; r < 8; ++r) {
                float sv = (mf > 0.0f) ? sc[r] * SCORE_SCALE : -1e9f;
                float tm = sv;                           // row-max within 16-lane half
                tm = fmaxf(tm, __shfl_xor(tm, 1, 32));
                tm = fmaxf(tm, __shfl_xor(tm, 2, 32));
                tm = fmaxf(tm, __shfl_xor(tm, 4, 32));
                tm = fmaxf(tm, __shfl_xor(tm, 8, 32));
                const float mn = fmaxf(m[r], tm);
                const float al = __expf(m[r] - mn);
                const float p  = __expf(sv - mn);
                l[r]  = l[r]  * al + p;                  // per-lane partials
                a0[r] = a0[r] * al + p * xv;
                a1[r] = a1[r] * al + p * mf;
                m[r]  = mn;
            }
        }
    }
#pragma unroll
    for (int r = 0; r < 8; ++r) {
        float ls = l[r], s0v = a0[r], s1v = a1[r];
#pragma unroll
        for (int d = 1; d <= 8; d <<= 1) {
            ls  += __shfl_xor(ls, d, 32);
            s0v += __shfl_xor(s0v, d, 32);
            s1v += __shfl_xor(s1v, d, 32);
        }
        if (lm == r) {                   // slot r owns rows r (lo half) / r+8 (hi half)
            const int q = qt * 16 + r + 8 * hi;
            const float inv = 1.0f / ls;
            float* o = concat + ((size_t)(b * LQ + q)) * 8 + hh * 2;
            o[0] = s0v * inv;
            o[1] = s1v * inv;
        }
    }
}

// =====================================================================
// K4: x_gates = concat @ W2.T + b2     (8192 x 8) -> (8192 x 384)
// =====================================================================
__global__ void xgates_kernel(const float* __restrict__ concat, const float* __restrict__ W2,
                              const float* __restrict__ b2, float* __restrict__ xg) {
    const int idx = blockIdx.x * blockDim.x + threadIdx.x;
    if (idx >= BS * LQ * 3 * NH) return;
    const int rg = idx / (3 * NH);
    const int g  = idx - rg * (3 * NH);
    const float* cr = concat + (size_t)rg * 8;
    const float* wr = W2 + (size_t)g * 8;
    float acc = b2[g];
#pragma unroll
    for (int c = 0; c < 8; ++c) acc = fmaf(cr[c], wr[c], acc);
    xg[idx] = acc;
}

// =====================================================================
// K5: GRU scan — one block per batch (independent), thread j = hidden unit
// =====================================================================
__global__ void __launch_bounds__(128) gru_kernel(const float* __restrict__ xg, const float* __restrict__ whh,
                           const float* __restrict__ bhh, float* __restrict__ gout) {
    const int b = blockIdx.x, j = threadIdx.x;
    __shared__ __align__(16) float h[NH];
    h[j] = 0.0f;
    __syncthreads();
    const float bh0 = bhh[j], bh1 = bhh[NH + j], bh2 = bhh[2 * NH + j];
    const float4* w0 = (const float4*)(whh + (size_t)j * NH);
    const float4* w1 = (const float4*)(whh + (size_t)(NH + j) * NH);
    const float4* w2 = (const float4*)(whh + (size_t)(2 * NH + j) * NH);
    for (int t = 0; t < LQ; ++t) {
        float rh = bh0, zh = bh1, nh = bh2;
        const float4* h4 = (const float4*)h;
#pragma unroll 8
        for (int k = 0; k < NH / 4; ++k) {
            const float4 hv = h4[k];
            const float4 wa = w0[k], wb = w1[k], wc = w2[k];
            rh += wa.x * hv.x + wa.y * hv.y + wa.z * hv.z + wa.w * hv.w;
            zh += wb.x * hv.x + wb.y * hv.y + wb.z * hv.z + wb.w * hv.w;
            nh += wc.x * hv.x + wc.y * hv.y + wc.z * hv.z + wc.w * hv.w;
        }
        const float* xr = xg + ((size_t)(b * LQ + t)) * (3 * NH);
        const float r = sigmoidf_(xr[j] + rh);
        const float z = sigmoidf_(xr[NH + j] + zh);
        const float n = tanhf(xr[2 * NH + j] + r * nh);
        const float hn = (1.0f - z) * n + z * h[j];
        __syncthreads();
        h[j] = hn;
        gout[((size_t)(b * LQ + t)) * NH + j] = hn;
        __syncthreads();
    }
}

// =====================================================================
// K6: attention pooling + 2x MLP regression heads -> T10/M0 in d_out tail
// =====================================================================
__global__ void pool_kernel(const float* __restrict__ gout,
                            const float* __restrict__ swT, const float* __restrict__ swM,
                            const float* __restrict__ tw1, const float* __restrict__ tb1,
                            const float* __restrict__ tw2, const float* __restrict__ tb2,
                            const float* __restrict__ tw3, const float* __restrict__ tb3,
                            const float* __restrict__ mw1, const float* __restrict__ mb1,
                            const float* __restrict__ mw2, const float* __restrict__ mb2,
                            const float* __restrict__ mw3, const float* __restrict__ mb3,
                            float* __restrict__ d_out) {
    const int b = blockIdx.x, tid = threadIdx.x;
    __shared__ float lT[LQ], lM[LQ], wT[LQ], wM[LQ], hT[NH], hM[NH];
    __shared__ float h1T[REGH], h1M[REGH], h2T[REGH], h2M[REGH];

    if (tid < LQ) {
        const float* orow = gout + ((size_t)(b * LQ + tid)) * NH;
        float aT = 0.0f, aM = 0.0f;
        for (int j = 0; j < NH; ++j) { aT = fmaf(orow[j], swT[j], aT); aM = fmaf(orow[j], swM[j], aM); }
        lT[tid] = aT; lM[tid] = aM;
    }
    __syncthreads();
    if (tid < LQ) {
        float mT = lT[0], mM = lM[0];
        for (int t = 1; t < LQ; ++t) { mT = fmaxf(mT, lT[t]); mM = fmaxf(mM, lM[t]); }
        float sT = 0.0f, sM = 0.0f;
        for (int t = 0; t < LQ; ++t) { sT += __expf(lT[t] - mT); sM += __expf(lM[t] - mM); }
        wT[tid] = __expf(lT[tid] - mT) / sT;
        wM[tid] = __expf(lM[tid] - mM) / sM;
    }
    __syncthreads();
    if (tid < NH) {
        float aT = 0.0f, aM = 0.0f;
        for (int t = 0; t < LQ; ++t) {
            const float o = gout[((size_t)(b * LQ + t)) * NH + tid];
            aT = fmaf(o, wT[t], aT); aM = fmaf(o, wM[t], aM);
        }
        hT[tid] = aT; hM[tid] = aM;
    }
    __syncthreads();
    if (tid < REGH) {
        float a1 = tb1[tid], a2 = mb1[tid];
        for (int j = 0; j < NH; ++j) {
            a1 = fmaf(tw1[(size_t)tid * NH + j], hT[j], a1);
            a2 = fmaf(mw1[(size_t)tid * NH + j], hM[j], a2);
        }
        h1T[tid] = gelu_exact(a1); h1M[tid] = gelu_exact(a2);
    }
    __syncthreads();
    if (tid < REGH) {
        float a1 = tb2[tid], a2 = mb2[tid];
        for (int j = 0; j < REGH; ++j) {
            a1 = fmaf(tw2[(size_t)tid * REGH + j], h1T[j], a1);
            a2 = fmaf(mw2[(size_t)tid * REGH + j], h1M[j], a2);
        }
        h2T[tid] = gelu_exact(a1); h2M[tid] = gelu_exact(a2);
    }
    __syncthreads();
    if (tid == 0) {
        float yT = tb3[0], yM = mb3[0];
        for (int j = 0; j < REGH; ++j) { yT = fmaf(tw3[j], h2T[j], yT); yM = fmaf(mw3[j], h2M[j], yM); }
        d_out[BS * SEQ + b]      = 0.1f + sigmoidf_(yT) * 4.9f;   // T10
        d_out[BS * SEQ + BS + b] = 0.1f + sigmoidf_(yM) * 9.9f;   // M0
    }
}

// =====================================================================
// K7: signal equation  X_out = (1-E) sin(fa) / (1 - cos(fa) E) * M0
// =====================================================================
__global__ void signal_kernel(const float* __restrict__ fa_vals, const float* __restrict__ TR_vals,
                              float* __restrict__ d_out) {
    const int idx = blockIdx.x * blockDim.x + threadIdx.x;
    if (idx >= BS * SEQ) return;
    const int b = idx >> 11;
    const float T10 = d_out[BS * SEQ + b];
    const float M0  = d_out[BS * SEQ + BS + b];
    const float fa = fa_vals[idx];
    const float E = __expf(-TR_vals[idx] / T10);
    float s, c;
    __sincosf(fa, &s, &c);
    d_out[idx] = (1.0f - E) * s / (1.0f - c * E) * M0;
}

// =====================================================================
extern "C" void kernel_launch(void* const* d_in, const int* in_sizes, int n_in,
                              void* d_out, int out_size, void* d_ws, size_t ws_size,
                              hipStream_t stream) {
    const float* X_fa     = (const float*)d_in[0];
    const float* fa_vals  = (const float*)d_in[1];
    const unsigned char* fa_mask = (const unsigned char*)d_in[2];  // jnp.bool_ -> 1 byte
    // d_in[3] = fa_len (unused by the reference computation)
    const float* TR_vals  = (const float*)d_in[4];
    const float* te_lin_w = (const float*)d_in[5];
    const float* te_lin_b = (const float*)d_in[6];
    const float* te_per_w = (const float*)d_in[7];
    const float* te_per_b = (const float*)d_in[8];
    const float* q_w = (const float*)d_in[9],  *q_b = (const float*)d_in[10];
    const float* k_w = (const float*)d_in[11], *k_b = (const float*)d_in[12];
    const float* aow = (const float*)d_in[13], *aob = (const float*)d_in[14];
    const float* wih = (const float*)d_in[15], *whh = (const float*)d_in[16];
    const float* bih = (const float*)d_in[17], *bhh = (const float*)d_in[18];
    const float* swT = (const float*)d_in[19], *swM = (const float*)d_in[20];
    const float* tw1 = (const float*)d_in[21], *tb1 = (const float*)d_in[22];
    const float* tw2 = (const float*)d_in[23], *tb2 = (const float*)d_in[24];
    const float* tw3 = (const float*)d_in[25], *tb3 = (const float*)d_in[26];
    const float* mw1 = (const float*)d_in[27], *mb1 = (const float*)d_in[28];
    const float* mw2 = (const float*)d_in[29], *mb2 = (const float*)d_in[30];
    const float* mw3 = (const float*)d_in[31], *mb3 = (const float*)d_in[32];
    float* out = (float*)d_out;

    uint8_t* ws = (uint8_t*)d_ws;
    size_t o = 0;
    auto carve = [&](size_t bytes) { uint8_t* p = ws + o; o = (o + bytes + 255) & ~(size_t)255; return p; };
    _Float16* q16   = (_Float16*)carve((size_t)NHEADS * LQ * DK * 2);          // 32 KB
    _Float16* k16   = (_Float16*)carve((size_t)BS * SEQ * DT * 2);             // 32 MB
    float* concat   = (float*)carve((size_t)BS * LQ * 8 * 4);                  // 256 KB
    float* W2       = (float*)carve((size_t)3 * NH * 8 * 4);
    float* b2       = (float*)carve((size_t)3 * NH * 4);
    float* xg       = (float*)carve((size_t)BS * LQ * 3 * NH * 4);             // 12.6 MB
    float* gout     = (float*)carve((size_t)BS * LQ * NH * 4);                 // 4 MB
    (void)ws_size; (void)in_sizes; (void)n_in; (void)out_size;

    qprep_kernel<<<LQ, DT, 0, stream>>>(te_lin_w, te_lin_b, te_per_w, te_per_b, q_w, q_b, q16);
    fusew2_kernel<<<3, 128, 0, stream>>>(wih, aow, aob, bih, W2, b2);
    kproj_kernel<<<BS * SEQ / 64, 256, 0, stream>>>(fa_vals, te_lin_w, te_lin_b, te_per_w, te_per_b,
                                                    k_w, k_b, k16);
    attn_kernel<<<BS * NHEADS, 256, 0, stream>>>(q16, k16, X_fa, fa_mask, concat);
    {
        const int total = BS * LQ * 3 * NH;
        xgates_kernel<<<(total + 255) / 256, 256, 0, stream>>>(concat, W2, b2, xg);
    }
    gru_kernel<<<BS, NH, 0, stream>>>(xg, whh, bhh, gout);
    pool_kernel<<<BS, 256, 0, stream>>>(gout, swT, swM, tw1, tb1, tw2, tb2, tw3, tb3,
                                        mw1, mb1, mw2, mb2, mw3, mb3, out);
    signal_kernel<<<(BS * SEQ + 255) / 256, 256, 0, stream>>>(fa_vals, TR_vals, out);
}
